// MplugOwlVisionAttention_1726576854686
// MI455X (gfx1250) — compile-verified, matrix operand order checked by reading, not run on hardware
//
#include <hip/hip_runtime.h>
#include <hip/hip_bf16.h>

// ---------------------------------------------------------------------------
// Types for CDNA5 WMMA (wave32): v_wmma_f32_16x16x32_bf16
// ---------------------------------------------------------------------------
typedef __attribute__((ext_vector_type(16))) __bf16 bf16x16;
typedef __attribute__((ext_vector_type(8)))  __bf16 bf16x8;
typedef __attribute__((ext_vector_type(8)))  float  f32x8;

#define HID   1024
#define HEADS 16
#define HD    64
#define BSZ   8
#define SEQ   1024
#define SCALE 0.125f  // 64^-0.5

__device__ __forceinline__ f32x8 zero8() {
  f32x8 z = {0.f, 0.f, 0.f, 0.f, 0.f, 0.f, 0.f, 0.f};
  return z;
}

__device__ __forceinline__ f32x8 wmma_bf16(bf16x16 a, bf16x16 b, f32x8 c) {
  // D = A(16x32 bf16) x B(32x16 bf16) + C(16x16 f32)
  return __builtin_amdgcn_wmma_f32_16x16x32_bf16(
      /*neg_a=*/false, a, /*neg_b=*/false, b,
      /*c_mod=*/(short)0, c, /*reuse_a=*/false, /*reuse_b=*/false);
}

// Load a WMMA A or B operand from a row-major matrix where the operand's
// per-lane row (lane&15) is a contiguous run of K-elements:
//   lane holds row (lane&15); halves 0..7 -> k = kb+0..7, halves 8..15 -> k = kb+16..23
//   where kb = (lane>=16) ? 8 : 0.   (ISA 7.12.2, 16-bit A 16x32 layout; B mirrors it.)
__device__ __forceinline__ bf16x16 load_op(const __bf16* base, int ldr, int lane) {
  const __bf16* p = base + (size_t)(lane & 15) * (size_t)ldr + (size_t)((lane >> 4) << 3);
  bf16x8 lo = *reinterpret_cast<const bf16x8*>(p);
  bf16x8 hi = *reinterpret_cast<const bf16x8*>(p + 16);
  return __builtin_shufflevector(lo, hi, 0, 1, 2, 3, 4, 5, 6, 7,
                                         8, 9, 10, 11, 12, 13, 14, 15);
}

// Row-wise reductions across the 16 lanes that share a C/D-layout row group.
// xor masks 1..8 never cross the lane<16 / lane>=16 boundary.
__device__ __forceinline__ float rowmax16(float t) {
#pragma unroll
  for (int off = 1; off <= 8; off <<= 1) t = fmaxf(t, __shfl_xor(t, off, 32));
  return t;
}
__device__ __forceinline__ float rowsum16(float t) {
#pragma unroll
  for (int off = 1; off <= 8; off <<= 1) t += __shfl_xor(t, off, 32);
  return t;
}

// ---------------------------------------------------------------------------
// Pass 0: f32 -> bf16 conversion
// ---------------------------------------------------------------------------
__global__ void cvt_f32_to_bf16(const float* __restrict__ in,
                                __bf16* __restrict__ out, int n) {
  int i = blockIdx.x * blockDim.x + threadIdx.x;
  int stride = gridDim.x * blockDim.x;
  for (; i < n; i += stride) out[i] = (__bf16)in[i];
}

// ---------------------------------------------------------------------------
// Pass 1: QKV GEMM.  Y[8192,3072] = Xbf @ Wqkv^T + b, scattered to:
//   Q,K : [b,h,s,64] bf16 ; V stored transposed Vt : [b,h,64,s] bf16
// Column n decomposes as n = h*192 + which*64 + d.
// One wave computes a 16x64 tile (4 N-subtiles), K-loop 1024/32.
// ---------------------------------------------------------------------------
__global__ __launch_bounds__(256)
void qkv_gemm(const __bf16* __restrict__ X, const __bf16* __restrict__ W,
              const float* __restrict__ bias,
              __bf16* __restrict__ Qo, __bf16* __restrict__ Ko,
              __bf16* __restrict__ Vt) {
  int lane = threadIdx.x & 31;
  int wv = threadIdx.x >> 5;
  int wid = blockIdx.x * 8 + wv;      // 0..24575
  int mt = wid & 511;                 // 512 M-tiles of 16
  int nt = wid >> 9;                  // 48 N-tiles of 64
  int m0 = mt * 16;
  int n0 = nt * 64;

  const __bf16* Abase = X + (size_t)m0 * HID;
  f32x8 acc0 = zero8(), acc1 = zero8(), acc2 = zero8(), acc3 = zero8();

  for (int kk = 0; kk < HID; kk += 32) {
    if (kk + 32 < HID) {
      __builtin_prefetch(Abase + kk + 32, 0, 3);
      __builtin_prefetch(W + (size_t)n0 * HID + kk + 32, 0, 3);
    }
    bf16x16 a = load_op(Abase + kk, HID, lane);
    acc0 = wmma_bf16(a, load_op(W + (size_t)(n0 +  0) * HID + kk, HID, lane), acc0);
    acc1 = wmma_bf16(a, load_op(W + (size_t)(n0 + 16) * HID + kk, HID, lane), acc1);
    acc2 = wmma_bf16(a, load_op(W + (size_t)(n0 + 32) * HID + kk, HID, lane), acc2);
    acc3 = wmma_bf16(a, load_op(W + (size_t)(n0 + 48) * HID + kk, HID, lane), acc3);
  }

  int which = nt % 3;
  int h = nt / 3;
  int b = m0 >> 10;
  int s0 = m0 & 1023;
  int bh = b * HEADS + h;

  f32x8 accs[4] = {acc0, acc1, acc2, acc3};
#pragma unroll
  for (int j = 0; j < 4; ++j) {
    int col = j * 16 + (lane & 15);                 // d in 0..63
    float bv = bias[n0 + col];
#pragma unroll
    for (int vg = 0; vg < 8; ++vg) {
      int row = vg + ((lane >> 4) << 3);            // M in 0..15
      float v = accs[j][vg] + bv;
      if (which == 2) {
        Vt[((size_t)bh * HD + col) * SEQ + (s0 + row)] = (__bf16)v;
      } else {
        __bf16* dst = (which == 0) ? Qo : Ko;
        dst[((size_t)bh * SEQ + (s0 + row)) * HD + col] = (__bf16)v;
      }
    }
  }
}

// ---------------------------------------------------------------------------
// Pass 2: flash attention.  One wave = one (b,h) x 16-row Q tile.
// Online softmax over 32-key blocks; P transposed C/D->A layout via LDS.
// ctx written bf16 to Ctx[b, s, h*64+d].
// ---------------------------------------------------------------------------
__global__ __launch_bounds__(256)
void attn_flash(const __bf16* __restrict__ Q, const __bf16* __restrict__ K,
                const __bf16* __restrict__ Vt, __bf16* __restrict__ Ctx) {
  __shared__ __align__(16) __bf16 ldsP[8][16][32];   // per-wave 16x32 P tile
  int lane = threadIdx.x & 31;
  int wv = threadIdx.x >> 5;
  int wid = blockIdx.x * 8 + wv;      // 0..8191
  int bh = wid >> 6;                  // 0..127
  int qt = wid & 63;                  // 0..63
  int b = bh >> 4, h = bh & 15;

  const __bf16* Qb = Q  + (size_t)bh * SEQ * HD + (size_t)qt * 16 * HD;
  const __bf16* Kb = K  + (size_t)bh * SEQ * HD;
  const __bf16* Vb = Vt + (size_t)bh * HD * SEQ;

  bf16x16 aq0 = load_op(Qb,      HD, lane);   // d = 0..31
  bf16x16 aq1 = load_op(Qb + 32, HD, lane);   // d = 32..63

  f32x8 acc0 = zero8(), acc1 = zero8(), acc2 = zero8(), acc3 = zero8();
  float m_r[8], l_r[8];
#pragma unroll
  for (int j = 0; j < 8; ++j) { m_r[j] = -__builtin_inff(); l_r[j] = 0.f; }

  for (int k0 = 0; k0 < SEQ; k0 += 32) {
    // ---- scores: two 16x16 tiles (keys k0..k0+15 and k0+16..k0+31) ----
    f32x8 S0 = zero8(), S1 = zero8();
    S0 = wmma_bf16(aq0, load_op(Kb + (size_t)k0 * HD,            HD, lane), S0);
    S0 = wmma_bf16(aq1, load_op(Kb + (size_t)k0 * HD + 32,       HD, lane), S0);
    S1 = wmma_bf16(aq0, load_op(Kb + (size_t)(k0 + 16) * HD,     HD, lane), S1);
    S1 = wmma_bf16(aq1, load_op(Kb + (size_t)(k0 + 16) * HD + 32, HD, lane), S1);

    // ---- online softmax ----
    float corr[8];
#pragma unroll
    for (int j = 0; j < 8; ++j) {
      float s0 = S0[j] * SCALE, s1 = S1[j] * SCALE;
      S0[j] = s0; S1[j] = s1;
      float mn = fmaxf(m_r[j], rowmax16(fmaxf(s0, s1)));
      corr[j] = __expf(m_r[j] - mn);
      m_r[j] = mn;
    }
#pragma unroll
    for (int j = 0; j < 8; ++j) {
      float p0 = __expf(S0[j] - m_r[j]);
      float p1 = __expf(S1[j] - m_r[j]);
      l_r[j] = l_r[j] * corr[j] + rowsum16(p0 + p1);
      int row = j + ((lane >> 4) << 3);
      int col = lane & 15;
      ldsP[wv][row][col]      = (__bf16)p0;
      ldsP[wv][row][col + 16] = (__bf16)p1;
      acc0[j] *= corr[j]; acc1[j] *= corr[j];
      acc2[j] *= corr[j]; acc3[j] *= corr[j];
    }
    // LDS RAW across lanes within the wave: split-counter wait (CDNA5)
    asm volatile("s_wait_dscnt 0" ::: "memory");
    __builtin_amdgcn_wave_barrier();

    bf16x16 aP = load_op(&ldsP[wv][0][0], 32, lane);   // P as 16x32 A operand
    // V^T rows are d-columns: contiguous loads, k pattern along the row
    acc0 = wmma_bf16(aP, load_op(Vb + (size_t)( 0) * SEQ + k0, SEQ, lane), acc0);
    acc1 = wmma_bf16(aP, load_op(Vb + (size_t)(16) * SEQ + k0, SEQ, lane), acc1);
    acc2 = wmma_bf16(aP, load_op(Vb + (size_t)(32) * SEQ + k0, SEQ, lane), acc2);
    acc3 = wmma_bf16(aP, load_op(Vb + (size_t)(48) * SEQ + k0, SEQ, lane), acc3);
    __builtin_amdgcn_wave_barrier();
  }

  __bf16* out = Ctx + ((size_t)(b * SEQ + qt * 16)) * HID + h * HD;
#pragma unroll
  for (int j = 0; j < 8; ++j) {
    int row = j + ((lane >> 4) << 3);
    int col = lane & 15;
    float inv = 1.0f / l_r[j];
    out[(size_t)row * HID + col     ] = (__bf16)(acc0[j] * inv);
    out[(size_t)row * HID + col + 16] = (__bf16)(acc1[j] * inv);
    out[(size_t)row * HID + col + 32] = (__bf16)(acc2[j] * inv);
    out[(size_t)row * HID + col + 48] = (__bf16)(acc3[j] * inv);
  }
}

// ---------------------------------------------------------------------------
// Pass 3: output projection, f32 result + bias straight to d_out.
// ---------------------------------------------------------------------------
__global__ __launch_bounds__(256)
void out_gemm(const __bf16* __restrict__ Ctx, const __bf16* __restrict__ W,
              const float* __restrict__ bias, float* __restrict__ Out) {
  int lane = threadIdx.x & 31;
  int wv = threadIdx.x >> 5;
  int wid = blockIdx.x * 8 + wv;      // 0..8191
  int mt = wid & 511;
  int nt = wid >> 9;                  // 0..15
  int m0 = mt * 16;
  int n0 = nt * 64;

  const __bf16* Abase = Ctx + (size_t)m0 * HID;
  f32x8 acc0 = zero8(), acc1 = zero8(), acc2 = zero8(), acc3 = zero8();

  for (int kk = 0; kk < HID; kk += 32) {
    if (kk + 32 < HID) {
      __builtin_prefetch(Abase + kk + 32, 0, 3);
      __builtin_prefetch(W + (size_t)n0 * HID + kk + 32, 0, 3);
    }
    bf16x16 a = load_op(Abase + kk, HID, lane);
    acc0 = wmma_bf16(a, load_op(W + (size_t)(n0 +  0) * HID + kk, HID, lane), acc0);
    acc1 = wmma_bf16(a, load_op(W + (size_t)(n0 + 16) * HID + kk, HID, lane), acc1);
    acc2 = wmma_bf16(a, load_op(W + (size_t)(n0 + 32) * HID + kk, HID, lane), acc2);
    acc3 = wmma_bf16(a, load_op(W + (size_t)(n0 + 48) * HID + kk, HID, lane), acc3);
  }

  f32x8 accs[4] = {acc0, acc1, acc2, acc3};
#pragma unroll
  for (int j = 0; j < 4; ++j) {
    int col = n0 + j * 16 + (lane & 15);
    float bv = bias[col];
#pragma unroll
    for (int vg = 0; vg < 8; ++vg) {
      int row = m0 + vg + ((lane >> 4) << 3);
      Out[(size_t)row * HID + col] = accs[j][vg] + bv;
    }
  }
}

// ---------------------------------------------------------------------------
// Host-side launch
// ---------------------------------------------------------------------------
extern "C" void kernel_launch(void* const* d_in, const int* in_sizes, int n_in,
                              void* d_out, int out_size, void* d_ws, size_t ws_size,
                              hipStream_t stream) {
  const float* X    = (const float*)d_in[0];   // [8,1024,1024]
  const float* Wqkv = (const float*)d_in[1];   // [3072,1024]
  const float* bqkv = (const float*)d_in[2];   // [3072]
  const float* Wout = (const float*)d_in[3];   // [1024,1024]
  const float* bout = (const float*)d_in[4];   // [1024]
  float* out = (float*)d_out;                  // [8,1024,1024] f32

  char* ws = (char*)d_ws;
  const size_t SZ_X    = (size_t)BSZ * SEQ * HID * 2;       // 16 MiB
  const size_t SZ_WQKV = (size_t)3 * HID * HID * 2;         //  6 MiB
  const size_t SZ_WOUT = (size_t)HID * HID * 2;             //  2 MiB
  const size_t SZ_QKV  = (size_t)BSZ * HEADS * SEQ * HD * 2;// 16 MiB each

  __bf16* Xbf    = (__bf16*)(ws);
  __bf16* Wqkvbf = (__bf16*)(ws + SZ_X);
  __bf16* Woutbf = (__bf16*)(ws + SZ_X + SZ_WQKV);
  __bf16* Qb     = (__bf16*)(ws + SZ_X + SZ_WQKV + SZ_WOUT);
  __bf16* Kb     = (__bf16*)(ws + SZ_X + SZ_WQKV + SZ_WOUT + SZ_QKV);
  __bf16* Vtb    = (__bf16*)(ws + SZ_X + SZ_WQKV + SZ_WOUT + 2 * SZ_QKV);
  __bf16* Ctx    = (__bf16*)(ws + SZ_X + SZ_WQKV + SZ_WOUT + 3 * SZ_QKV);

  cvt_f32_to_bf16<<<2048, 256, 0, stream>>>(X,    Xbf,    BSZ * SEQ * HID);
  cvt_f32_to_bf16<<<1024, 256, 0, stream>>>(Wqkv, Wqkvbf, 3 * HID * HID);
  cvt_f32_to_bf16<<<512,  256, 0, stream>>>(Wout, Woutbf, HID * HID);

  // 512 M-tiles x 48 N-tiles = 24576 waves / 8 waves-per-block
  qkv_gemm<<<3072, 256, 0, stream>>>(Xbf, Wqkvbf, bqkv, Qb, Kb, Vtb);
  // 128 (b,h) x 64 q-tiles = 8192 waves / 8
  attn_flash<<<1024, 256, 0, stream>>>(Qb, Kb, Vtb, Ctx);
  // 512 M-tiles x 16 N-tiles = 8192 waves / 8
  out_gemm<<<1024, 256, 0, stream>>>(Ctx, Woutbf, bout, out);
}
